// ConvolutionalModel_44555990729204
// MI455X (gfx1250) — matrix operand order repaired and checked by reading, not compiled, not run on hardware
//
#include <hip/hip_runtime.h>
#include <hip/hip_bf16.h>

typedef __attribute__((ext_vector_type(16))) _Float16 v16h;
typedef __attribute__((ext_vector_type(8)))  float    v8f;
typedef __attribute__((ext_vector_type(4)))  float    f4;

#define NWAVE 8                       // waves per block (8 * 32 = 256 threads)

__device__ __forceinline__ v8f wmma16(v16h a, v16h b, v8f c) {
  // D = A(16x32 f16) * B(32x16 f16) + C(16x16 f32)
  return __builtin_amdgcn_wmma_f32_16x16x32_f16(false, a, false, b, (short)0, c,
                                                false, false);
}

// x:[16,1,1024,1024] f32. K=4 windows, 16 consecutive windows (along W) per wave.
// Chain: win(16) -[We]-> 64 -[Wr,relu]-> 64 -[Ws]-> 16, gated by relu(win.Wa+ba),
// out = x + gated update.
__global__ __launch_bounds__(NWAVE * 32)
void conv_recursive_fused(const float* __restrict__ x,
                          const float* __restrict__ Wa, const float* __restrict__ ba,
                          const float* __restrict__ We, const float* __restrict__ be,
                          const float* __restrict__ Wr, const float* __restrict__ br,
                          const float* __restrict__ Ws, const float* __restrict__ bs,
                          float* __restrict__ out, int nTiles)
{
  // Per-wave scratch: no cross-wave sharing -> zero barriers.
  __shared__ __align__(16) _Float16 sAct[NWAVE][16 * 64];  // activation staging (f16)
  __shared__ __align__(16) float    sRep[NWAVE][16 * 16];  // gated replica (f32)
  __shared__               float    sAttn[NWAVE][16];

  const int lane = threadIdx.x & 31;
  const int wv   = threadIdx.x >> 5;
  const int hi   = lane >> 4;   // half-wave: selects K-groups / window rows
  const int col  = lane & 15;   // N index (feature col / window id / window elem)

  // ---------------- weights -> WMMA B-layout registers (reused for all tiles) ---
  // B 32x16 layout: lane<16 holds K={0..7,16..23}, lane>=16 holds K={8..15,24..31}; N=lane%16.
  v16h bWe[4];                                   // GEMM1: K=0..15 real, 16..31 pad
#pragma unroll
  for (int nt = 0; nt < 4; ++nt) {
    v16h b = {};
    {
      const int n = nt * 16 + col;
#pragma unroll
      for (int p = 0; p < 8; ++p) b[p] = (_Float16)We[(hi * 8 + p) * 64 + n];
    }
    bWe[nt] = b;
  }
  v16h bWr[2][4];                                // GEMM2: K=64 in two 32-chunks
#pragma unroll
  for (int kc = 0; kc < 2; ++kc)
#pragma unroll
    for (int nt = 0; nt < 4; ++nt) {
      const int n = nt * 16 + col;
      const int kb = kc * 32 + hi * 8;
      v16h b;
#pragma unroll
      for (int p = 0; p < 16; ++p) b[p] = (_Float16)Wr[(kb + p + (p >= 8 ? 8 : 0)) * 64 + n];
      bWr[kc][nt] = b;
    }
  v16h bWs[2];                                   // GEMM3: N=16
#pragma unroll
  for (int kc = 0; kc < 2; ++kc) {
    const int kb = kc * 32 + hi * 8;
    v16h b;
#pragma unroll
    for (int p = 0; p < 16; ++p) b[p] = (_Float16)Ws[(kb + p + (p >= 8 ? 8 : 0)) * 16 + col];
    bWs[kc] = b;
  }
  float beV[4], brV[4];
#pragma unroll
  for (int nt = 0; nt < 4; ++nt) {
    beV[nt] = be[nt * 16 + col];
    brV[nt] = br[nt * 16 + col];
  }
  const float bsV = bs[col];
  const float baV = ba[0];
  float waL[8];
#pragma unroll
  for (int j = 0; j < 8; ++j) waL[j] = Wa[hi * 8 + j];

  // ---------------- grid-stride loop over 16-window tiles ----------------------
  const int gw     = blockIdx.x * NWAVE + wv;
  const int stride = gridDim.x * NWAVE;
  for (int t = gw; t < nTiles; t += stride) {
    const int b_ = t >> 12;          // 4096 tiles per batch image
    const int hh = (t >> 4) & 255;   // window row
    const int wg = t & 15;           // group of 16 windows along W
    const int base = b_ * 1048576 + hh * 4096 + wg * 64;
    const int a0 = base + (hi * 2) * 1024 + col * 4;  // rows hi*2, hi*2+1
    const int a1 = a0 + 1024;

    const f4 x0 = *(const f4*)(x + a0);   // window `col`, row hi*2   (k = hi*8+0..3)
    const f4 x1 = *(const f4*)(x + a1);   // window `col`, row hi*2+1 (k = hi*8+4..7)

    // attention gate: relu(win . Wa + ba)
    float part = x0.x * waL[0] + x0.y * waL[1] + x0.z * waL[2] + x0.w * waL[3] +
                 x1.x * waL[4] + x1.y * waL[5] + x1.z * waL[6] + x1.w * waL[7];
    part += __shfl_xor(part, 16, 32);
    float attn = part + baV;
    attn = attn > 0.f ? attn : 0.f;
    if (hi == 0) sAttn[wv][col] = attn;

    // A1 16x32 f16: lane<16 -> K=0..7 in halves 0..7 (16..23 pad0);
    //               lane>=16 -> K=8..15 in halves 0..7 (24..31 pad0).
    v16h aWin = {};
    aWin[0] = (_Float16)x0.x; aWin[1] = (_Float16)x0.y;
    aWin[2] = (_Float16)x0.z; aWin[3] = (_Float16)x0.w;
    aWin[4] = (_Float16)x1.x; aWin[5] = (_Float16)x1.y;
    aWin[6] = (_Float16)x1.z; aWin[7] = (_Float16)x1.w;

    // -------- GEMM1: expanded[16x64] = win @ We + be --------
    v8f accE[4];
#pragma unroll
    for (int nt = 0; nt < 4; ++nt) {
      v8f c;
#pragma unroll
      for (int i = 0; i < 8; ++i) c[i] = beV[nt];
      accE[nt] = wmma16(aWin, bWe[nt], c);
    }
    // stage expanded (D layout: lane holds M=v+8*hi, N=nt*16+col) as row-major f16
#pragma unroll
    for (int nt = 0; nt < 4; ++nt)
#pragma unroll
      for (int v = 0; v < 8; ++v)
        sAct[wv][(v + 8 * hi) * 64 + nt * 16 + col] = (_Float16)accE[nt][v];

    // -------- GEMM2: rec[16x64] = relu(expanded @ Wr + br) --------
    v8f accR[4];
#pragma unroll
    for (int nt = 0; nt < 4; ++nt) {
      v8f c;
#pragma unroll
      for (int i = 0; i < 8; ++i) c[i] = brV[nt];
      accR[nt] = c;
    }
#pragma unroll
    for (int kc = 0; kc < 2; ++kc) {
      // A-layout reload: lane<16 K=kc*32+{0..7,16..23}; lane>=16 +8.
      const _Float16* rp = &sAct[wv][col * 64];
      v16h a2;
#pragma unroll
      for (int j = 0; j < 8; ++j) a2[j]     = rp[kc * 32 + hi * 8 + j];
#pragma unroll
      for (int j = 0; j < 8; ++j) a2[8 + j] = rp[kc * 32 + 16 + hi * 8 + j];
#pragma unroll
      for (int nt = 0; nt < 4; ++nt) accR[nt] = wmma16(a2, bWr[kc][nt], accR[nt]);
    }
#pragma unroll
    for (int nt = 0; nt < 4; ++nt) {
#pragma unroll
      for (int v = 0; v < 8; ++v) {
        float r = accR[nt][v];
        r = r > 0.f ? r : 0.f;
        sAct[wv][(v + 8 * hi) * 64 + nt * 16 + col] = (_Float16)r;  // stage rec
      }
    }

    // -------- GEMM3: rep[16x16] = rec @ Ws + bs --------
    v8f accS;
#pragma unroll
    for (int i = 0; i < 8; ++i) accS[i] = bsV;
#pragma unroll
    for (int kc = 0; kc < 2; ++kc) {
      const _Float16* rp = &sAct[wv][col * 64];
      v16h a3;
#pragma unroll
      for (int j = 0; j < 8; ++j) a3[j]     = rp[kc * 32 + hi * 8 + j];
#pragma unroll
      for (int j = 0; j < 8; ++j) a3[8 + j] = rp[kc * 32 + 16 + hi * 8 + j];
      accS = wmma16(a3, bWs[kc], accS);
    }

    // gate by attn and re-swizzle for coalesced store: sRep[m][k]
#pragma unroll
    for (int v = 0; v < 8; ++v) {
      const int M = v + 8 * hi;
      sRep[wv][M * 16 + col] = accS[v] * sAttn[wv][M];
    }
    const float* rr = &sRep[wv][col * 16 + hi * 8];
    const f4 u0 = *(const f4*)(rr);
    const f4 u1 = *(const f4*)(rr + 4);
    f4 o0, o1;
    o0.x = x0.x + u0.x; o0.y = x0.y + u0.y; o0.z = x0.z + u0.z; o0.w = x0.w + u0.w;
    o1.x = x1.x + u1.x; o1.y = x1.y + u1.y; o1.z = x1.z + u1.z; o1.w = x1.w + u1.w;
    *(f4*)(out + a0) = o0;
    *(f4*)(out + a1) = o1;
  }
}

extern "C" void kernel_launch(void* const* d_in, const int* in_sizes, int n_in,
                              void* d_out, int out_size, void* d_ws, size_t ws_size,
                              hipStream_t stream) {
  const float* x  = (const float*)d_in[0];
  const float* Wa = (const float*)d_in[1];
  const float* ba = (const float*)d_in[2];
  const float* We = (const float*)d_in[3];
  const float* be = (const float*)d_in[4];
  const float* Wr = (const float*)d_in[5];
  const float* br = (const float*)d_in[6];
  const float* Ws = (const float*)d_in[7];
  const float* bs = (const float*)d_in[8];
  float* out = (float*)d_out;

  // 16 windows per tile, 16 elems per window -> tiles = elems / 256
  const int nTiles = in_sizes[0] / 256;   // 65536 for 16x1x1024x1024
  const int blocks = 2048;                // 16384 waves, 4 tiles each (grid-stride)
  conv_recursive_fused<<<blocks, NWAVE * 32, 0, stream>>>(
      x, Wa, ba, We, be, Wr, br, Ws, bs, out, nTiles);
}
